// RecurrentTransformerBlock_8615704396127
// MI455X (gfx1250) — compile-verified, matrix operand order checked by reading, not run on hardware
//
#include <hip/hip_runtime.h>
#include <hip/hip_bf16.h>
#include <math.h>

#define DIM 2048
#define HID 5324
#define HID_PAD 5376   // 42 * 128, zero padded
#define BATCH 16384

typedef __bf16 bf16_t;
typedef bf16_t v16bf __attribute__((ext_vector_type(16)));
typedef bf16_t v8bf  __attribute__((ext_vector_type(8)));
typedef float  v8f   __attribute__((ext_vector_type(8)));
typedef unsigned int v4u __attribute__((ext_vector_type(4)));
typedef int v4i __attribute__((ext_vector_type(4)));
typedef int v8i __attribute__((ext_vector_type(8)));

#define BM 128
#define BN 128
#define BK 32
#define LDSS 40   // LDS row stride in bf16 elements: 64B row + 16B TDM pad = 80B (16B aligned)

// ---------------------------------------------------------------------------
// Tensor Data Mover: async copy of a 128x32 bf16 tile (row-major, row stride K
// elements) from global memory into LDS at byte offset lds_off, with hardware
// row padding of 16B per 64B row (matches LDSS=40).
// D# layout per CDNA5 ISA ch.8 (CDNA5 ISA).
// ---------------------------------------------------------------------------
__device__ __forceinline__ void tdm_load_tile(const bf16_t* gptr, unsigned lds_off,
                                              int K, int rows)
{
    const unsigned long long ga = (unsigned long long)(size_t)gptr;
    v4u g0 = (v4u)0u;
    g0[0] = 1u;                                         // count=1 (valid user descriptor)
    g0[1] = lds_off;                                    // lds_addr (bytes)
    g0[2] = (unsigned)(ga & 0xFFFFFFFFu);               // global_addr[31:0]
    g0[3] = (unsigned)((ga >> 32) & 0x01FFFFFFu)        // global_addr[56:32]
          | (2u << 30);                                 // type=2 ("image")

    v8i g1 = (v8i)0;
    // data_size=1 (2 bytes), pad_enable, pad_interval=3 (16 DWORDs = 64B),
    // pad_amount=3 (4 DWORDs = 16B); workgroup_mask=0 (not in cluster)
    g1[0] = (1 << 16) | (1 << 20) | (3 << 22) | (3 << 25);
    g1[1] = (K & 0xFFFF) << 16;                         // tensor_dim0[15:0] (bits 63:48)
    g1[2] = ((unsigned)K >> 16)                         // tensor_dim0[31:16] (bits 79:64)
          | ((rows & 0xFFFF) << 16);                    // tensor_dim1[15:0] (bits 95:80)
    g1[3] = ((unsigned)rows >> 16)                      // tensor_dim1[31:16] (bits 111:96)
          | (BK << 16);                                 // tile_dim0=32 elems (bits 127:112)
    g1[4] = BM;                                         // tile_dim1=128 rows (bits 143:128); tile_dim2=0
    g1[5] = K;                                          // tensor_dim0_stride[31:0] (bits 191:160)
    g1[6] = 0;                                          // stride[47:32]=0; tensor_dim1_stride unused (2D)
    g1[7] = 0;

    const v4i z4 = (v4i)0;
#if __clang_major__ >= 23
    const v8i z8 = (v8i)0;
    __builtin_amdgcn_tensor_load_to_lds(g0, g1, z4, z4, z8, 0);
#else
    __builtin_amdgcn_tensor_load_to_lds(g0, g1, z4, z4, 0);
#endif
}

// Build a v16bf WMMA fragment from LDS following the CDNA5 16-bit A/B layout:
// lanes 0-15: elems 0..7 = K[khalf..khalf+7], elems 8..15 = K[khalf+16..khalf+23]
__device__ __forceinline__ v16bf load_frag(const bf16_t* p, int row, int khalf) {
    const bf16_t* r = p + row * LDSS + khalf;
    v8bf lo = *(const v8bf*)(r);
    v8bf hi = *(const v8bf*)(r + 16);
    return __builtin_shufflevector(lo, hi, 0,1,2,3,4,5,6,7,8,9,10,11,12,13,14,15);
}

// MODE 0: outb(bf16) = A*B0^T + bias          (v projection)
// MODE 1: res(f32)  += A*B0^T + bias          (attn out + residual, in-place)
// MODE 2: outb(bf16) = silu(A*B0^T)*(A*B1^T)  (gated MLP front)
// MODE 3: outf(f32)  = res + A*B0^T           (MLP down proj + residual)
template<int MODE>
__global__ __launch_bounds__(256, 1) void gemm_wmma(
    const bf16_t* __restrict__ A,  const bf16_t* __restrict__ B0,
    const bf16_t* __restrict__ B1, const float* __restrict__ bias,
    float* __restrict__ res, float* __restrict__ outf,
    bf16_t* __restrict__ outb, int K)
{
    __shared__ alignas(16) bf16_t sA[2 * BM * LDSS];
    __shared__ alignas(16) bf16_t sB[2 * BN * LDSS];
    __shared__ alignas(16) bf16_t sB2[(MODE == 2) ? 2 * BN * LDSS : 16];

    const int tid  = threadIdx.x;
    const int wave = tid >> 5;
    const int lane = tid & 31;
    const int hh   = lane >> 4;      // half-wave select
    const int l16  = lane & 15;
    const int wm   = wave >> 2;      // 0..1  -> 64 rows each
    const int wn   = wave & 3;       // 0..3  -> 32 cols each
    const int bn   = blockIdx.x * BN;
    const int bm   = blockIdx.y * BM;
    const int N    = gridDim.x * BN;

    const bf16_t* Ab  = A  + (size_t)bm * K;
    const bf16_t* Bb  = B0 + (size_t)bn * K;
    const bf16_t* B2b = (MODE == 2) ? (B1 + (size_t)bn * K) : nullptr;

    const unsigned sAoff  = (unsigned)(size_t)(void*)sA;
    const unsigned sBoff  = (unsigned)(size_t)(void*)sB;
    const unsigned sB2off = (unsigned)(size_t)(void*)sB2;
    const unsigned bufBytes = BM * LDSS * 2;   // 10240 B per buffer

    v8f acc[4][2];
    v8f acc2[4][2];
    #pragma unroll
    for (int i = 0; i < 4; ++i)
        #pragma unroll
        for (int j = 0; j < 2; ++j) { acc[i][j] = (v8f)0.0f; acc2[i][j] = (v8f)0.0f; }

    // Prologue: wave 0 launches the TDM DMAs for buffer 0.
    if (wave == 0) {
        tdm_load_tile(Ab, sAoff, K, BATCH);
        tdm_load_tile(Bb, sBoff, K, N);
        if (MODE == 2) tdm_load_tile(B2b, sB2off, K, N);
    }

    const int NT = K / BK;
    for (int kt = 0; kt < NT; ++kt) {
        const int cur = kt & 1;

        // Wave 0 waits for the DMA filling `cur`; the barrier then both
        // publishes "tile ready" to all waves and guarantees everyone is done
        // reading buffer cur^1 before the next DMA overwrites it.
        if (wave == 0) __builtin_amdgcn_s_wait_tensorcnt(0);
        __syncthreads();
        if (kt + 1 < NT && wave == 0) {
            const int kk = (kt + 1) * BK;
            const unsigned nb = (cur ^ 1) * bufBytes;
            tdm_load_tile(Ab + kk, sAoff + nb, K, BATCH);
            tdm_load_tile(Bb + kk, sBoff + nb, K, N);
            if (MODE == 2) tdm_load_tile(B2b + kk, sB2off + nb, K, N);
        }

        const bf16_t* pA = sA + cur * BM * LDSS;
        const bf16_t* pB = sB + cur * BN * LDSS;

        v16bf af[4], bfr[2];
        #pragma unroll
        for (int i = 0; i < 4; ++i) af[i]  = load_frag(pA, wm * 64 + i * 16 + l16, hh * 8);
        #pragma unroll
        for (int j = 0; j < 2; ++j) bfr[j] = load_frag(pB, wn * 32 + j * 16 + l16, hh * 8);

        #pragma unroll
        for (int i = 0; i < 4; ++i)
            #pragma unroll
            for (int j = 0; j < 2; ++j)
                acc[i][j] = __builtin_amdgcn_wmma_f32_16x16x32_bf16(
                    false, af[i], false, bfr[j], (short)0, acc[i][j], false, false);

        if (MODE == 2) {
            const bf16_t* pC = sB2 + cur * BN * LDSS;
            v16bf cfr[2];
            #pragma unroll
            for (int j = 0; j < 2; ++j) cfr[j] = load_frag(pC, wn * 32 + j * 16 + l16, hh * 8);
            #pragma unroll
            for (int i = 0; i < 4; ++i)
                #pragma unroll
                for (int j = 0; j < 2; ++j)
                    acc2[i][j] = __builtin_amdgcn_wmma_f32_16x16x32_bf16(
                        false, af[i], false, cfr[j], (short)0, acc2[i][j], false, false);
        }
    }

    // Epilogue. C layout: VGPR r, lanes 0-15 -> M=r, lanes 16-31 -> M=r+8; N = lane&15.
    #pragma unroll
    for (int i = 0; i < 4; ++i) {
        #pragma unroll
        for (int j = 0; j < 2; ++j) {
            const int n = bn + wn * 32 + j * 16 + l16;
            float bv = 0.0f;
            if (MODE == 0 || MODE == 1) bv = bias[n];
            #pragma unroll
            for (int r = 0; r < 8; ++r) {
                const int m = bm + wm * 64 + i * 16 + hh * 8 + r;
                const size_t idx = (size_t)m * N + n;
                const float val = acc[i][j][r];
                if (MODE == 0) {
                    outb[idx] = (bf16_t)(val + bv);
                } else if (MODE == 1) {
                    res[idx] = res[idx] + val + bv;
                } else if (MODE == 2) {
                    const float a = val;
                    const float b = acc2[i][j][r];
                    const float s = a / (1.0f + __expf(-a));
                    outb[idx] = (bf16_t)(s * b);
                } else {
                    outf[idx] = res[idx] + val;
                }
            }
        }
    }
}

// h = x + state ; hn = rmsnorm(h) * g1 (bf16)
__global__ __launch_bounds__(256) void prep_kernel(
    const float* __restrict__ x, const float* __restrict__ st,
    const float* __restrict__ g, float* __restrict__ h, bf16_t* __restrict__ hn)
{
    const size_t base = (size_t)blockIdx.x * DIM;
    float v[8];
    float ss = 0.0f;
    #pragma unroll
    for (int i = 0; i < 8; ++i) {
        const int idx = threadIdx.x + i * 256;
        const float t = x[base + idx] + st[base + idx];
        v[i] = t; ss += t * t;
        h[base + idx] = t;
    }
    #pragma unroll
    for (int o = 16; o > 0; o >>= 1) ss += __shfl_xor(ss, o, 32);
    __shared__ float sred[8];
    if ((threadIdx.x & 31) == 0) sred[threadIdx.x >> 5] = ss;
    __syncthreads();
    float tot = 0.0f;
    #pragma unroll
    for (int i = 0; i < 8; ++i) tot += sred[i];
    const float scale = 0.02209708691207961f / fmaxf(sqrtf(tot), 1e-12f); // DIM^-0.5 / max(||h||,eps)
    #pragma unroll
    for (int i = 0; i < 8; ++i) {
        const int idx = threadIdx.x + i * 256;
        hn[base + idx] = (bf16_t)(v[i] * scale * g[idx]);
    }
}

// hn = rmsnorm(h) * g (bf16)
__global__ __launch_bounds__(256) void norm_kernel(
    const float* __restrict__ h, const float* __restrict__ g, bf16_t* __restrict__ hn)
{
    const size_t base = (size_t)blockIdx.x * DIM;
    float v[8];
    float ss = 0.0f;
    #pragma unroll
    for (int i = 0; i < 8; ++i) {
        const int idx = threadIdx.x + i * 256;
        const float t = h[base + idx];
        v[i] = t; ss += t * t;
    }
    #pragma unroll
    for (int o = 16; o > 0; o >>= 1) ss += __shfl_xor(ss, o, 32);
    __shared__ float sred[8];
    if ((threadIdx.x & 31) == 0) sred[threadIdx.x >> 5] = ss;
    __syncthreads();
    float tot = 0.0f;
    #pragma unroll
    for (int i = 0; i < 8; ++i) tot += sred[i];
    const float scale = 0.02209708691207961f / fmaxf(sqrtf(tot), 1e-12f);
    #pragma unroll
    for (int i = 0; i < 8; ++i) {
        const int idx = threadIdx.x + i * 256;
        hn[base + idx] = (bf16_t)(v[i] * scale * g[idx]);
    }
}

// fp32 -> bf16 with zero padding outside [src_rows, src_cols]
__global__ __launch_bounds__(256) void cvt_kernel(
    const float* __restrict__ src, bf16_t* __restrict__ dst,
    int src_rows, int src_cols, int dst_cols)
{
    const int c = blockIdx.x * 256 + threadIdx.x;
    const int r = blockIdx.y;
    const float v = (r < src_rows && c < src_cols) ? src[(size_t)r * src_cols + c] : 0.0f;
    dst[(size_t)r * dst_cols + c] = (bf16_t)v;
}

extern "C" void kernel_launch(void* const* d_in, const int* in_sizes, int n_in,
                              void* d_out, int out_size, void* d_ws, size_t ws_size,
                              hipStream_t stream)
{
    const float* x          = (const float*)d_in[0];
    const float* state      = (const float*)d_in[1];
    const float* g1         = (const float*)d_in[2];
    const float* g2         = (const float*)d_in[3];
    const float* in_proj_w  = (const float*)d_in[4];
    const float* in_proj_b  = (const float*)d_in[5];
    const float* out_proj_w = (const float*)d_in[6];
    const float* out_proj_b = (const float*)d_in[7];
    const float* w1         = (const float*)d_in[8];
    const float* w2         = (const float*)d_in[9];
    const float* w3         = (const float*)d_in[10];
    float* out = (float*)d_out;

    char* ws = (char*)d_ws;
    size_t off = 0;
    auto alloc = [&](size_t bytes) -> void* {
        void* p = ws + off;
        off += (bytes + 255) & ~(size_t)255;
        return p;
    };
    float*  h    = (float*) alloc((size_t)BATCH * DIM * 4);      // residual, updated in place
    bf16_t* hn   = (bf16_t*)alloc((size_t)BATCH * DIM * 2);      // reused for hn2
    bf16_t* vbuf = (bf16_t*)alloc((size_t)BATCH * DIM * 2);
    bf16_t* tbuf = (bf16_t*)alloc((size_t)BATCH * HID_PAD * 2);
    bf16_t* wvb  = (bf16_t*)alloc((size_t)DIM * DIM * 2);
    bf16_t* owb  = (bf16_t*)alloc((size_t)DIM * DIM * 2);
    bf16_t* w1b  = (bf16_t*)alloc((size_t)HID_PAD * DIM * 2);
    bf16_t* w2b  = (bf16_t*)alloc((size_t)HID_PAD * DIM * 2);
    bf16_t* w3b  = (bf16_t*)alloc((size_t)DIM * HID_PAD * 2);
    (void)ws_size; (void)in_sizes; (void)n_in; (void)out_size;

    // Weight conversions (bandwidth-trivial vs the GEMMs)
    cvt_kernel<<<dim3(DIM / 256, DIM),      256, 0, stream>>>(in_proj_w + 2 * DIM * DIM, wvb, DIM, DIM, DIM);
    cvt_kernel<<<dim3(DIM / 256, DIM),      256, 0, stream>>>(out_proj_w, owb, DIM, DIM, DIM);
    cvt_kernel<<<dim3(DIM / 256, HID_PAD),  256, 0, stream>>>(w1, w1b, HID, DIM, DIM);
    cvt_kernel<<<dim3(DIM / 256, HID_PAD),  256, 0, stream>>>(w2, w2b, HID, DIM, DIM);
    cvt_kernel<<<dim3(HID_PAD / 256, DIM),  256, 0, stream>>>(w3, w3b, DIM, HID, HID_PAD);

    // h = x + state ; hn = rmsnorm(h)*g1
    prep_kernel<<<BATCH, 256, 0, stream>>>(x, state, g1, h, hn);

    // v = hn @ Wv^T + bv
    gemm_wmma<0><<<dim3(DIM / BN, BATCH / BM), 256, 0, stream>>>(
        hn, wvb, nullptr, in_proj_b + 2 * DIM, nullptr, nullptr, vbuf, DIM);

    // h += v @ Wo^T + bo
    gemm_wmma<1><<<dim3(DIM / BN, BATCH / BM), 256, 0, stream>>>(
        vbuf, owb, nullptr, out_proj_b, h, nullptr, nullptr, DIM);

    // hn2 = rmsnorm(h)*g2
    norm_kernel<<<BATCH, 256, 0, stream>>>(h, g2, hn);

    // t = silu(hn2 @ W1^T) * (hn2 @ W2^T)   (padded N; pad cols compute to 0)
    gemm_wmma<2><<<dim3(HID_PAD / BN, BATCH / BM), 256, 0, stream>>>(
        hn, w1b, w2b, nullptr, nullptr, nullptr, tbuf, DIM);

    // out = h + t @ W3^T   (padded K contributes 0)
    gemm_wmma<3><<<dim3(DIM / BN, BATCH / BM), 256, 0, stream>>>(
        tbuf, w3b, nullptr, nullptr, h, out, nullptr, HID_PAD);
}
